// JunmaiLayer_32126355374637
// MI455X (gfx1250) — matrix operand order repaired and checked by reading, not compiled, not run on hardware
//
#include <hip/hip_runtime.h>

// ---------- types ----------
typedef __attribute__((ext_vector_type(16))) __bf16 v16bf;
typedef __attribute__((ext_vector_type(2)))  __bf16 bf16x2;
typedef __attribute__((ext_vector_type(8)))  float  v8f;
typedef __attribute__((ext_vector_type(4)))  int    v4i_;

union BF16x16 { unsigned int u[8]; v16bf v; };

#if __has_builtin(__builtin_amdgcn_global_load_async_to_lds_b128)
#define HAVE_ASYNC_LDS 1
typedef __attribute__((address_space(1))) v4i_* gv4p;   // global v4i*
typedef __attribute__((address_space(3))) v4i_* lv4p;   // LDS v4i*
#endif

// ---------- helpers (native bf16 converts -> v_cvt_pk_bf16_f32 etc.) ----------
__device__ __forceinline__ unsigned short f2bf(float f) {
    union { __bf16 h; unsigned short s; } r;
    r.h = (__bf16)f;
    return r.s;
}
__device__ __forceinline__ unsigned int pack2bf(float a, float b) {
    union { bf16x2 v; unsigned int u; } r;
    r.v[0] = (__bf16)a; r.v[1] = (__bf16)b;
    return r.u;
}
// packed bf16 pair multiply (two bf16 in each u32) -> v_pk_mul_bf16
__device__ __forceinline__ unsigned int pkmul(unsigned int a, unsigned int b) {
    union U { unsigned int u; bf16x2 v; };
    U x, y, r;
    x.u = a; y.u = b;
    r.v = x.v * y.v;
    return r.u;
}
__device__ __forceinline__ void wait_async0() {
#if defined(HAVE_ASYNC_LDS)
#if __has_builtin(__builtin_amdgcn_s_wait_asynccnt)
    __builtin_amdgcn_s_wait_asynccnt(0);
#else
    asm volatile("s_wait_asynccnt 0x0" ::: "memory");
#endif
#endif
}

// =====================================================================
// Kernel 1: hc = h @ W_coef + b_coef ; route into h0,h2 (f32 [n][k][o])
// and h1,h3 (bf16, [k][o][m]) for WMMA-B staging.
// =====================================================================
__global__ __launch_bounds__(256) void junmai_coef_kernel(
    const float* __restrict__ h, const float* __restrict__ W_coef,
    const float* __restrict__ b_coef,
    float* __restrict__ h0f, float* __restrict__ h2f,
    unsigned short* __restrict__ h1bf, unsigned short* __restrict__ h3bf)
{
    __shared__ float hrow[64];
    const int n = blockIdx.x >> 4;
    const int j = ((blockIdx.x & 15) << 8) + threadIdx.x;   // 0..4095
    if (threadIdx.x < 64) hrow[threadIdx.x] = h[n * 64 + threadIdx.x];
    __syncthreads();
    float acc = b_coef[j];
#pragma unroll 8
    for (int r = 0; r < 64; ++r) acc += hrow[r] * W_coef[r * 4096 + j];
    const int k = j >> 7, rem = j & 127, sel = rem >> 5, o = rem & 31;
    if (sel == 0)      h0f[(n * 32 + k) * 32 + o] = acc;
    else if (sel == 1) h1bf[(k * 32 + o) * 1024 + n] = f2bf(acc);
    else if (sel == 2) h2f[(n * 32 + k) * 32 + o] = acc;
    else               h3bf[(k * 32 + o) * 1024 + n] = f2bf(acc);
}

// =====================================================================
// Kernel 2: pairwise fused kernel.
// grid = (64 n-tiles, 2 k-groups), block = 512 threads (16 waves).
// Wave w owns k = kgrp*16 + w for the whole m loop.
// Per 16x32 pair tile: compute diff+smear -> LDS; basis = smear@W_basis
// via WMMA; A_c = basis * diff_c (v_pk_mul_bf16); accumulate
// P1 (A @ h1) and T (A @ h3) with v_wmma_f32_16x16x32_bf16.
// h1/h3 tiles staged with GLOBAL_LOAD_ASYNC_TO_LDS when available.
// =====================================================================
__global__ __launch_bounds__(512) void junmai_pair_kernel(
    const float* __restrict__ x,
    const float* __restrict__ W_basis, const float* __restrict__ b_basis,
    const float* __restrict__ means,   const float* __restrict__ betas,
    const unsigned short* __restrict__ h1bf, const unsigned short* __restrict__ h3bf,
    const float* __restrict__ h2f,
    float* __restrict__ P1, float* __restrict__ P2)
{
    __shared__ float meansS[32], betasS[32];
    __shared__ unsigned short WbT[16 * 32];          // [k_local][rbf]
    __shared__ unsigned short smearS[512 * 32];      // [pair][rbf]
    __shared__ unsigned short basisS[16 * 16 * 32];  // [k_local][n][m]
    __shared__ unsigned short diffS[3 * 16 * 32];    // [c][n][m]
    __shared__ unsigned short h1S[16 * 32 * 32];     // [k_local][o][m]
    __shared__ unsigned short h3S[16 * 32 * 32];

    const int tid   = threadIdx.x;
    const int lane  = tid & 31;
    const int wave  = tid >> 5;           // 0..15
    const int ntile = blockIdx.x;         // 0..63
    const int kgrp  = blockIdx.y;         // 0..1
    const int k     = kgrp * 16 + wave;   // this wave's k

    if (tid < 32) { meansS[tid] = means[tid]; betasS[tid] = betas[tid]; }
    {   // stage W_basis^T (only the 16 k-columns this workgroup needs)
        const int kl = tid >> 5, r = tid & 31;
        WbT[kl * 32 + r] = f2bf(W_basis[r * 32 + (kgrp * 16 + kl)]);
    }

    const v8f vzero = {0.f, 0.f, 0.f, 0.f, 0.f, 0.f, 0.f, 0.f};
    v8f p1acc[3][2], tacc[3][2];
#pragma unroll
    for (int c = 0; c < 3; ++c)
#pragma unroll
        for (int oh = 0; oh < 2; ++oh) { p1acc[c][oh] = vzero; tacc[c][oh] = vzero; }

    // this thread's pair: p = tid, n fixed across the m loop
    const int nl = tid >> 5, ml = tid & 31;
    const int ng = ntile * 16 + nl;
    const float xn0 = x[ng * 3 + 0], xn1 = x[ng * 3 + 1], xn2 = x[ng * 3 + 2];

    // per-thread staging row for h1/h3 ([k][o][m])
    const int skk = tid >> 5, so = tid & 31;
    const unsigned int* g1base = (const unsigned int*)h1bf + (((kgrp * 16 + skk) * 32 + so) * 512);
    const unsigned int* g3base = (const unsigned int*)h3bf + (((kgrp * 16 + skk) * 32 + so) * 512);
    unsigned int* l1 = (unsigned int*)&h1S[(skk * 32 + so) * 32];
    unsigned int* l3 = (unsigned int*)&h3S[(skk * 32 + so) * 32];

    __syncthreads();

    for (int mt = 0; mt < 32; ++mt) {
        // ---------------- stage phase ----------------
        {   // h1/h3 tiles: async copy global -> LDS (falls back to VGPR copy)
            const unsigned int* g1 = g1base + mt * 16;
            const unsigned int* g3 = g3base + mt * 16;
#if defined(HAVE_ASYNC_LDS)
#pragma unroll
            for (int jj = 0; jj < 16; jj += 4) {
                __builtin_amdgcn_global_load_async_to_lds_b128(
                    (gv4p)(g1 + jj), (lv4p)(l1 + jj), 0, 0);
                __builtin_amdgcn_global_load_async_to_lds_b128(
                    (gv4p)(g3 + jj), (lv4p)(l3 + jj), 0, 0);
            }
#else
#pragma unroll
            for (int jj = 0; jj < 16; jj += 4) {
                *(uint4*)(l1 + jj) = *(const uint4*)(g1 + jj);
                *(uint4*)(l3 + jj) = *(const uint4*)(g3 + jj);
            }
#endif
        }
        {   // pair geometry + smear (overlaps with async DMA above)
            const int mg = mt * 32 + ml;
            const float dx = xn0 - x[mg * 3 + 0];
            const float dy = xn1 - x[mg * 3 + 1];
            const float dz = xn2 - x[mg * 3 + 2];
            const float nsq  = dx * dx + dy * dy + dz * dz + 1e-6f;
            const float dist = sqrtf(nsq);
            const float inv  = 1.0f / nsq;
            diffS[(0 * 16 + nl) * 32 + ml] = f2bf(dx * inv);
            diffS[(1 * 16 + nl) * 32 + ml] = f2bf(dy * inv);
            diffS[(2 * 16 + nl) * 32 + ml] = f2bf(dz * inv);
            const float cutoff = (dist < 5.0f)
                ? 0.5f * (__cosf(dist * 0.62831853071795864f) + 1.0f) : 0.0f;
            const float e = __expf(-dist);   // alpha=1, CUT_LO=0
            unsigned int* srow = (unsigned int*)&smearS[tid * 32];
#pragma unroll
            for (int r = 0; r < 32; r += 2) {
                const float t0 = e - meansS[r];
                const float t1 = e - meansS[r + 1];
                const float s0 = cutoff * __expf(-betasS[r] * t0 * t0);
                const float s1 = cutoff * __expf(-betasS[r + 1] * t1 * t1);
                srow[r >> 1] = pack2bf(s0, s1);
            }
        }
        wait_async0();
        __syncthreads();

        // ---------------- basis = smear @ W_basis (WMMA) ----------------
#pragma unroll
        for (int i = 0; i < 2; ++i) {
            const int msub = wave * 2 + i;          // 0..31 pair-subtile
            BF16x16 A, Bm;
            {
                const int prow = msub * 16 + (lane & 15);
                const int rb = (lane < 16) ? 0 : 8;
                const unsigned int* sr = (const unsigned int*)&smearS[prow * 32];
#pragma unroll
                for (int v = 0; v < 4; ++v) {
                    A.u[v]     = sr[(rb + 2 * v) >> 1];
                    A.u[4 + v] = sr[(16 + rb + 2 * v) >> 1];
                }
            }
            {
                const int kl = lane & 15;
                const int rb = (lane < 16) ? 0 : 8;
                const unsigned int* wr = (const unsigned int*)&WbT[kl * 32];
#pragma unroll
                for (int v = 0; v < 4; ++v) {
                    Bm.u[v]     = wr[(rb + 2 * v) >> 1];
                    Bm.u[4 + v] = wr[(16 + rb + 2 * v) >> 1];
                }
            }
            v8f cacc = vzero;
            cacc = __builtin_amdgcn_wmma_f32_16x16x32_bf16(
                false, A.v, false, Bm.v, (short)0, cacc, false, false);
            const int kl = lane & 15;
            const float bb = b_basis[kgrp * 16 + kl];
            // rows prow0..prow0+7 are contiguous m within one n -> one b128 store
            const int prow0 = msub * 16 + ((lane < 16) ? 0 : 8);
            unsigned int pk[4];
#pragma unroll
            for (int v = 0; v < 4; ++v)
                pk[v] = pack2bf(cacc[2 * v] + bb, cacc[2 * v + 1] + bb);
            *(uint4*)&basisS[(kl * 16 + (prow0 >> 5)) * 32 + (prow0 & 31)] =
                *(const uint4*)pk;
        }
        __syncthreads();

        // ---------------- main accumulation (wave = one k) ----------------
        BF16x16 bas, bh1[2], bh3[2];
        {
            const int an = lane & 15;
            const int mb = (lane < 16) ? 0 : 8;
            const unsigned int* br = (const unsigned int*)&basisS[(wave * 16 + an) * 32];
#pragma unroll
            for (int v = 0; v < 4; ++v) {
                bas.u[v]     = br[(mb + 2 * v) >> 1];
                bas.u[4 + v] = br[(16 + mb + 2 * v) >> 1];
            }
        }
        {
            const int o0 = lane & 15;
            const int mb = (lane < 16) ? 0 : 8;
#pragma unroll
            for (int oh = 0; oh < 2; ++oh) {
                const unsigned int* r1 = (const unsigned int*)&h1S[(wave * 32 + oh * 16 + o0) * 32];
                const unsigned int* r3 = (const unsigned int*)&h3S[(wave * 32 + oh * 16 + o0) * 32];
#pragma unroll
                for (int v = 0; v < 4; ++v) {
                    bh1[oh].u[v]     = r1[(mb + 2 * v) >> 1];
                    bh1[oh].u[4 + v] = r1[(16 + mb + 2 * v) >> 1];
                    bh3[oh].u[v]     = r3[(mb + 2 * v) >> 1];
                    bh3[oh].u[4 + v] = r3[(16 + mb + 2 * v) >> 1];
                }
            }
        }
#pragma unroll
        for (int c = 0; c < 3; ++c) {
            BF16x16 Ac;
            const int an = lane & 15;
            const int mb = (lane < 16) ? 0 : 8;
            const unsigned int* dr = (const unsigned int*)&diffS[(c * 16 + an) * 32];
#pragma unroll
            for (int v = 0; v < 4; ++v) {
                Ac.u[v]     = pkmul(bas.u[v],     dr[(mb + 2 * v) >> 1]);
                Ac.u[4 + v] = pkmul(bas.u[4 + v], dr[(16 + mb + 2 * v) >> 1]);
            }
#pragma unroll
            for (int oh = 0; oh < 2; ++oh) {
                p1acc[c][oh] = __builtin_amdgcn_wmma_f32_16x16x32_bf16(
                    false, Ac.v, false, bh1[oh].v, (short)0, p1acc[c][oh], false, false);
                tacc[c][oh] = __builtin_amdgcn_wmma_f32_16x16x32_bf16(
                    false, Ac.v, false, bh3[oh].v, (short)0, tacc[c][oh], false, false);
            }
        }
        __syncthreads();
    }

    // ---------------- epilogue: write P1, reduce T*h2 into P2 ----------------
    {
        const int o0   = lane & 15;
        const int nsel = (lane < 16) ? 0 : 8;
        float h2v[2][8];
#pragma unroll
        for (int oh = 0; oh < 2; ++oh)
#pragma unroll
            for (int v = 0; v < 8; ++v) {
                const int n = ntile * 16 + v + nsel;
                h2v[oh][v] = h2f[(n * 32 + k) * 32 + (oh * 16 + o0)];
            }
#pragma unroll
        for (int c = 0; c < 3; ++c)
#pragma unroll
            for (int oh = 0; oh < 2; ++oh) {
                float psum = 0.f;
#pragma unroll
                for (int v = 0; v < 8; ++v) {
                    const int n = ntile * 16 + v + nsel;
                    const int o = oh * 16 + o0;
                    P1[((n * 32 + k) * 3 + c) * 32 + o] = p1acc[c][oh][v];
                    psum += tacc[c][oh][v] * h2v[oh][v];
                }
                atomicAdd(&P2[(k * 3 + c) * 32 + (oh * 16 + o0)], psum);
            }
    }
}

// =====================================================================
// Kernel 3: out[n,:] = silu( sum_{k,c} P1*P2*h0 ) @ W_sum + b_sum
// =====================================================================
__global__ __launch_bounds__(64) void junmai_out_kernel(
    const float* __restrict__ P1, const float* __restrict__ P2,
    const float* __restrict__ h0f,
    const float* __restrict__ W_sum, const float* __restrict__ b_sum,
    float* __restrict__ out)
{
    __shared__ float sh[32];
    const int n = blockIdx.x;
    const int t = threadIdx.x;
    if (t < 32) {
        float s = 0.f;
        for (int k = 0; k < 32; ++k) {
            float acc = 0.f;
#pragma unroll
            for (int c = 0; c < 3; ++c)
                acc += P1[((n * 32 + k) * 3 + c) * 32 + t] * P2[(k * 3 + c) * 32 + t];
            s += h0f[(n * 32 + k) * 32 + t] * acc;
        }
        sh[t] = s / (1.0f + __expf(-s));
    }
    __syncthreads();
    float acc = b_sum[t];
#pragma unroll 8
    for (int o = 0; o < 32; ++o) acc += sh[o] * W_sum[o * 64 + t];
    out[n * 64 + t] = acc;
}

// =====================================================================
extern "C" void kernel_launch(void* const* d_in, const int* in_sizes, int n_in,
                              void* d_out, int out_size, void* d_ws, size_t ws_size,
                              hipStream_t stream)
{
    const float* h       = (const float*)d_in[0];
    const float* x       = (const float*)d_in[1];
    const float* W_coef  = (const float*)d_in[2];
    const float* b_coef  = (const float*)d_in[3];
    const float* W_basis = (const float*)d_in[4];
    const float* b_basis = (const float*)d_in[5];
    const float* W_sum   = (const float*)d_in[6];
    const float* b_sum   = (const float*)d_in[7];
    const float* means   = (const float*)d_in[8];
    const float* betas   = (const float*)d_in[9];
    float* out = (float*)d_out;

    char* ws = (char*)d_ws;
    float*          h0f  = (float*)(ws + 0);                 // 4 MB
    float*          h2f  = (float*)(ws + 4194304);           // 4 MB
    unsigned short* h1bf = (unsigned short*)(ws + 8388608);  // 2 MB
    unsigned short* h3bf = (unsigned short*)(ws + 10485760); // 2 MB
    float*          P1   = (float*)(ws + 12582912);          // 12.58 MB
    float*          P2   = (float*)(ws + 25165824);          // 12 KB

    (void)hipMemsetAsync(P2, 0, 3072 * sizeof(float), stream);
    junmai_coef_kernel<<<16384, 256, 0, stream>>>(h, W_coef, b_coef, h0f, h2f, h1bf, h3bf);
    dim3 grid2(64, 2);
    junmai_pair_kernel<<<grid2, 512, 0, stream>>>(x, W_basis, b_basis, means, betas,
                                                  h1bf, h3bf, h2f, P1, P2);
    junmai_out_kernel<<<1024, 64, 0, stream>>>(P1, P2, h0f, W_sum, b_sum, out);
}
